// JoinAtention_9268539425554
// MI455X (gfx1250) — matrix operand order repaired and checked by reading, not compiled, not run on hardware
//
#include <hip/hip_runtime.h>
#include <hip/hip_bf16.h>

typedef __attribute__((ext_vector_type(16))) __bf16 v16bf;
typedef __attribute__((ext_vector_type(8)))  __bf16 v8bf;
typedef __attribute__((ext_vector_type(8)))  float  v8f;

namespace cfg {
constexpr int B = 2, T = 2048, MAXN = 64, DIM = 1024, H = 16, Dh = 64, R = 1;
constexpr int P = MAXN * R;          // 64
constexpr int S = P + T;             // 2112
constexpr int AD = H * Dh;           // 1024
// workspace layout (bytes)
constexpr size_t SZ_WQ = (size_t)3 * AD * DIM * 2;   // transposed qkv weights, bf16
constexpr size_t SZ_WO = (size_t)DIM * AD * 2;       // transposed out weights, bf16
constexpr size_t SZ_AN = (size_t)B * S * DIM * 2;    // normalized activations, bf16
constexpr size_t SZ_HB = (size_t)B * H * S * Dh * 2; // per-head q/k/v, bf16
constexpr size_t OFF_WQP = 0;
constexpr size_t OFF_WQX = OFF_WQP + SZ_WQ;
constexpr size_t OFF_WOP = OFF_WQX + SZ_WQ;
constexpr size_t OFF_WOX = OFF_WOP + SZ_WO;
constexpr size_t OFF_AN  = OFF_WOX + SZ_WO;
constexpr size_t OFF_QB  = OFF_AN + SZ_AN;
constexpr size_t OFF_KB  = OFF_QB + SZ_HB;
constexpr size_t OFF_VT  = OFF_KB + SZ_HB;
constexpr size_t OFF_AO  = OFF_VT + SZ_HB;
constexpr size_t OFF_LP  = OFF_AO + SZ_AN;
}

__device__ __forceinline__ v8f zero8() {
  v8f z;
#pragma unroll
  for (int i = 0; i < 8; ++i) z[i] = 0.0f;
  return z;
}

// Build a 16-element bf16 fragment from two 16-byte chunks (global or LDS).
__device__ __forceinline__ v16bf pack16(const __bf16* p0, const __bf16* p1) {
  v8bf a = *(const v8bf*)p0;
  v8bf b = *(const v8bf*)p1;
  v16bf r;
#pragma unroll
  for (int i = 0; i < 8; ++i) { r[i] = a[i]; r[i + 8] = b[i]; }
  return r;
}

__device__ __forceinline__ v8f wmma_bf16(v16bf a, v16bf b, v8f c) {
  return __builtin_amdgcn_wmma_f32_16x16x32_bf16(false, a, false, b, (short)0, c,
                                                 false, false);
}

// reductions within a 16-lane half of the wave32 (C/D row layout groups)
__device__ __forceinline__ float redmax16(float v) {
#pragma unroll
  for (int off = 1; off < 16; off <<= 1) v = fmaxf(v, __shfl_xor(v, off, 32));
  return v;
}
__device__ __forceinline__ float redsum16(float v) {
#pragma unroll
  for (int off = 1; off < 16; off <<= 1) v += __shfl_xor(v, off, 32);
  return v;
}

// Async copy of one 16B chunk global -> LDS (ASYNCcnt-tracked, CDNA5 path).
// Flat shared addresses carry the LDS byte offset in their low 32 bits.
__device__ __forceinline__ void async_load16(const __bf16* gsrc, __bf16* ldst) {
  unsigned lds_addr = (unsigned)(unsigned long long)ldst;
  asm volatile("global_load_async_to_lds_b128 %0, %1, off"
               :
               : "v"(lds_addr), "v"(gsrc)
               : "memory");
}

// ---------------- kernel 1: weight convert + transpose to bf16 ----------------
__global__ __launch_bounds__(256) void convert_weights(
    const float* wq_p, const float* wq_x, const float* wo_p, const float* wo_x,
    __bf16* tq_p, __bf16* tq_x, __bf16* to_p, __bf16* to_x) {
  using namespace cfg;
  size_t idx = (size_t)blockIdx.x * 256 + threadIdx.x;
  constexpr size_t NQ = (size_t)3 * AD * DIM;
  constexpr size_t NO = (size_t)DIM * AD;
  if (idx < NQ) {                                   // tq_p[n][k] = wq_p[k][n]
    size_t n = idx / DIM, k = idx % DIM;
    tq_p[idx] = (__bf16)wq_p[k * (size_t)(3 * AD) + n];
  } else if (idx < 2 * NQ) {
    size_t j = idx - NQ, n = j / DIM, k = j % DIM;
    tq_x[j] = (__bf16)wq_x[k * (size_t)(3 * AD) + n];
  } else if (idx < 2 * NQ + NO) {                   // to_p[n][k] = wo_p[k][n]
    size_t j = idx - 2 * NQ, n = j / AD, k = j % AD;
    to_p[j] = (__bf16)wo_p[k * (size_t)DIM + n];
  } else if (idx < 2 * NQ + 2 * NO) {
    size_t j = idx - 2 * NQ - NO, n = j / AD, k = j % AD;
    to_x[j] = (__bf16)wo_x[k * (size_t)DIM + n];
  }
}

// ---------------- kernel 2: region-local positions ----------------
__global__ void local_pos(const int* regions, int* lpos) {
  using namespace cfg;
  if (threadIdx.x != 0) return;
  int b = blockIdx.x;
  int prev = 0, run = 0;
  for (int t = 0; t < T; ++t) {
    int r = regions[b * T + t];
    if (r != prev) run = t;
    lpos[b * T + t] = (r > 0) ? (t - run + R) : 0;
    prev = r;
  }
}

// ---------------- kernel 3: token RMSNorm -> bf16 ----------------
__global__ __launch_bounds__(256) void rmsnorm_tokens(const float* pool, const float* x,
                                                      const float* pw, const float* xw,
                                                      __bf16* an) {
  using namespace cfg;
  int row = blockIdx.x;              // b*S + s
  int b = row / S, s = row % S;
  const float* src = (s < P) ? (pool + (size_t)(b * P + s) * DIM)
                             : (x + (size_t)(b * T + (s - P)) * DIM);
  const float* w = (s < P) ? pw : xw;
  float4 v = ((const float4*)src)[threadIdx.x];
  float ss = v.x * v.x + v.y * v.y + v.z * v.z + v.w * v.w;
#pragma unroll
  for (int off = 1; off < 32; off <<= 1) ss += __shfl_xor(ss, off, 32);
  __shared__ float buf[8];
  if ((threadIdx.x & 31) == 0) buf[threadIdx.x >> 5] = ss;
  __syncthreads();
  float total = 0.f;
#pragma unroll
  for (int i = 0; i < 8; ++i) total += buf[i];
  float sc = rsqrtf(total / (float)DIM + 1e-6f);
  float4 wv = ((const float4*)w)[threadIdx.x];
  __bf16* dst = an + (size_t)row * DIM + threadIdx.x * 4;
  dst[0] = (__bf16)(v.x * sc * wv.x);
  dst[1] = (__bf16)(v.y * sc * wv.y);
  dst[2] = (__bf16)(v.z * sc * wv.z);
  dst[3] = (__bf16)(v.w * sc * wv.w);
}

// ---------------- kernel 4: QKV GEMM + head RMSNorm + rotary, fused ----------------
// 4 waves/block: each wave owns one 16-token x 64-feature tile; the 64x32 weight
// slab per k-step is staged once per block into double-buffered LDS with async
// loads and consumed by all 4 waves.
__global__ __launch_bounds__(128) void qkv_fused(
    const __bf16* an, const __bf16* wq_p, const __bf16* wq_x,
    const float* bq_p, const float* bq_x,
    const float* qn_p, const float* kn_p, const float* qn_x, const float* kn_x,
    const int* lpos, __bf16* qh, __bf16* kh, __bf16* vt) {
  using namespace cfg;
  __shared__ __align__(16) __bf16 wbuf[2][64][40];   // 32 k + 8 pad per row
  const int tid = threadIdx.x;
  const int wave = tid >> 5;
  const int lane = tid & 31, g = (lane >> 4), ln = lane & 15;
  const int b = blockIdx.z;
  const int m0 = (blockIdx.y * 4 + wave) * 16;       // 4 m-tiles per block
  const int n0 = blockIdx.x * 64;
  const bool isPool = (m0 < P);                      // block-uniform (P==64)
  const __bf16* W = isPool ? wq_p : wq_x;
  const float* bias = isPool ? bq_p : bq_x;
  const __bf16* arow = an + (size_t)(b * S + m0 + ln) * DIM;

  auto stage = [&](int buf, int k0) {                // 64 rows x 32 k = 256 chunks
#pragma unroll
    for (int i = 0; i < 2; ++i) {
      int idx = tid + i * 128;
      int row = idx >> 2, c2 = idx & 3;
      async_load16(W + (size_t)(n0 + row) * DIM + k0 + c2 * 8,
                   &wbuf[buf][row][c2 * 8]);
    }
  };

  v8f acc[4];
#pragma unroll
  for (int c = 0; c < 4; ++c) acc[c] = zero8();

  stage(0, 0);                                       // ASYNCcnt = 2 per wave
  for (int k0 = 0; k0 < DIM; k0 += 32) {
    const int cur = (k0 >> 5) & 1;
    if (k0 + 32 < DIM) {
      stage(cur ^ 1, k0 + 32);                       // ASYNCcnt = 4
      asm volatile("s_wait_asynccnt 0x2" ::: "memory");  // current buffer done
    } else {
      asm volatile("s_wait_asynccnt 0x0" ::: "memory");
    }
    __syncthreads();                                 // all waves' chunks visible
    v16bf af = pack16(arow + k0 + g * 8, arow + k0 + 16 + g * 8);
#pragma unroll
    for (int c = 0; c < 4; ++c) {
      const __bf16* wr = &wbuf[cur][c * 16 + ln][8 * g];
      acc[c] = wmma_bf16(af, pack16(wr, wr + 16), acc[c]);
    }
    __syncthreads();                                 // done reading before overwrite
  }

#pragma unroll
  for (int c = 0; c < 4; ++c) {
    float bv = bias[n0 + c * 16 + ln];
#pragma unroll
    for (int r = 0; r < 8; ++r) acc[c][r] += bv;
  }
  const int section = n0 / AD;          // 0=q, 1=k, 2=v
  const int h = (n0 % AD) / Dh;
  if (section == 2) {                   // V: store transposed [B][H][Dh][S]
#pragma unroll
    for (int c = 0; c < 4; ++c) {
      int d = c * 16 + ln;
      __bf16* dst = vt + ((size_t)(b * H + h) * Dh + d) * S;
#pragma unroll
      for (int r = 0; r < 8; ++r) dst[m0 + r + 8 * g] = (__bf16)acc[c][r];
    }
    return;
  }
  // per-head RMSNorm over d=0..63 (row = C-layout row r+8g)
  const float* nw = (section == 0) ? (isPool ? qn_p : qn_x) : (isPool ? kn_p : kn_x);
  float inv_rms[8];
#pragma unroll
  for (int r = 0; r < 8; ++r) {
    float t = 0.f;
#pragma unroll
    for (int c = 0; c < 4; ++c) t += acc[c][r] * acc[c][r];
    t = redsum16(t);
    inv_rms[r] = rsqrtf(t / (float)Dh + 1e-6f);
  }
#pragma unroll
  for (int c = 0; c < 4; ++c) {
    float w = nw[c * 16 + ln];
#pragma unroll
    for (int r = 0; r < 8; ++r) acc[c][r] *= inv_rms[r] * w;
  }
  // rotary: d<32 uses global pos, d>=32 uses region-local pos
  float gp[8], rp[8];
#pragma unroll
  for (int r = 0; r < 8; ++r) {
    int s = m0 + r + 8 * g;
    gp[r] = (s < P) ? (float)s : (float)(s - P);
    rp[r] = (s < P) ? 0.f : (float)lpos[b * T + (s - P)];
  }
  const float lt = 9.210340371976184f / 16.f;  // ln(10000)/16
#pragma unroll
  for (int c = 0; c < 4; ++c) {
    int d = c * 16 + ln;
    int j = (d & 31) >> 1;
    float invf = __expf(-(float)j * lt);
    bool odd = (d & 1) != 0;
#pragma unroll
    for (int r = 0; r < 8; ++r) {
      float pos = (d < 32) ? gp[r] : rp[r];
      float sn, cs;
      __sincosf(pos * invf, &sn, &cs);
      float mine = acc[c][r];
      float other = __shfl_xor(mine, 1, 32);   // pair partner (d^1), same lane group
      acc[c][r] = odd ? (other * sn + mine * cs) : (mine * cs - other * sn);
    }
  }
  __bf16* base = ((section == 0) ? qh : kh) + (size_t)(b * H + h) * S * Dh;
#pragma unroll
  for (int c = 0; c < 4; ++c) {
#pragma unroll
    for (int r = 0; r < 8; ++r)
      base[(size_t)(m0 + r + 8 * g) * Dh + c * 16 + ln] = (__bf16)acc[c][r];
  }
}

// ---------------- kernel 5: flash attention, 16 query rows / wave ----------------
__global__ __launch_bounds__(32) void flash_attn(
    const __bf16* qh, const __bf16* kh, const __bf16* vt,
    const unsigned char* amask, __bf16* ao) {
  using namespace cfg;
  __shared__ __align__(16) __bf16 ptile[16][32];
  const int lane = threadIdx.x, g = lane >> 4, ln = lane & 15;
  const int m0 = blockIdx.x * 16, h = blockIdx.y, b = blockIdx.z;
  const __bf16* qrow = qh + ((size_t)(b * H + h) * S + m0 + ln) * Dh;
  v16bf qf[2];
#pragma unroll
  for (int t = 0; t < 2; ++t)
    qf[t] = pack16(qrow + t * 32 + g * 8, qrow + t * 32 + 16 + g * 8);
  const __bf16* kbase = kh + (size_t)(b * H + h) * S * Dh;
  const __bf16* vbase = vt + (size_t)(b * H + h) * Dh * S;
  const unsigned char* mrow = amask + (size_t)b * S * S;
  float mstate[8], lstate[8];
  v8f o[4];
#pragma unroll
  for (int r = 0; r < 8; ++r) { mstate[r] = -__builtin_inff(); lstate[r] = 0.f; }
#pragma unroll
  for (int c = 0; c < 4; ++c) o[c] = zero8();

  for (int j0 = 0; j0 < S; j0 += 32) {
    if (j0 + 32 < S) {   // prefetch next K/V tile cachelines (global_prefetch)
      __builtin_prefetch(kbase + (size_t)(j0 + 32 + lane) * Dh, 0, 3);
      __builtin_prefetch(vbase + (size_t)lane * S + j0 + 32, 0, 3);
      __builtin_prefetch(vbase + (size_t)(lane + 32) * S + j0 + 32, 0, 3);
    }
    v8f s0 = zero8(), s1 = zero8();
#pragma unroll
    for (int t = 0; t < 2; ++t) {                  // accumulate over d halves (K=64)
      const __bf16* k0p = kbase + (size_t)(j0 + ln) * Dh + t * 32 + 8 * g;
      s0 = wmma_bf16(qf[t], pack16(k0p, k0p + 16), s0);
      const __bf16* k1p = kbase + (size_t)(j0 + 16 + ln) * Dh + t * 32 + 8 * g;
      s1 = wmma_bf16(qf[t], pack16(k1p, k1p + 16), s1);
    }
    // online softmax update (row lives in one 16-lane group, one VGPR slot)
#pragma unroll
    for (int r = 0; r < 8; ++r) {
      const int q = m0 + r + 8 * g;
      float a0 = s0[r] * 0.125f;                   // 1/sqrt(64)
      float a1 = s1[r] * 0.125f;
      if (!mrow[(size_t)q * S + j0 + ln]) a0 = -1e30f;
      if (!mrow[(size_t)q * S + j0 + 16 + ln]) a1 = -1e30f;
      float mx = redmax16(fmaxf(a0, a1));
      float mn = fmaxf(mstate[r], mx);
      float alpha = __expf(mstate[r] - mn);
      float p0 = __expf(a0 - mn);
      float p1 = __expf(a1 - mn);
      float rs = redsum16(p0 + p1);
      lstate[r] = lstate[r] * alpha + rs;
      mstate[r] = mn;
#pragma unroll
      for (int c = 0; c < 4; ++c) o[c][r] *= alpha;
      ptile[r + 8 * g][ln] = (__bf16)p0;
      ptile[r + 8 * g][16 + ln] = (__bf16)p1;
    }
    __syncthreads();                               // C-layout -> A-layout via LDS
    v16bf pf = pack16(&ptile[ln][g * 8], &ptile[ln][16 + g * 8]);
#pragma unroll
    for (int c = 0; c < 4; ++c) {                  // O += P(16x32) @ V(32x64)
      const __bf16* vr = vbase + (size_t)(c * 16 + ln) * S + j0 + 8 * g;
      o[c] = wmma_bf16(pf, pack16(vr, vr + 16), o[c]);
    }
    __syncthreads();
  }
#pragma unroll
  for (int r = 0; r < 8; ++r) {
    const int q = m0 + r + 8 * g;
    float inv = 1.f / lstate[r];
#pragma unroll
    for (int c = 0; c < 4; ++c)
      ao[(size_t)(b * S + q) * AD + h * Dh + c * 16 + ln] = (__bf16)(o[c][r] * inv);
  }
}

// ---------------- kernel 6: output projection + bias + masks ----------------
__global__ __launch_bounds__(128) void out_proj(
    const __bf16* ao, const __bf16* wo_p, const __bf16* wo_x,
    const float* bo_p, const float* bo_x,
    const unsigned char* nmask, const unsigned char* tmask, float* out) {
  using namespace cfg;
  __shared__ __align__(16) __bf16 wbuf[2][64][40];
  const int tid = threadIdx.x;
  const int wave = tid >> 5;
  const int lane = tid & 31, g = (lane >> 4), ln = lane & 15;
  const int b = blockIdx.z;
  const int m0 = (blockIdx.y * 4 + wave) * 16;
  const int n0 = blockIdx.x * 64;
  const bool isPool = (m0 < P);                      // block-uniform
  const __bf16* W = isPool ? wo_p : wo_x;
  const float* bias = isPool ? bo_p : bo_x;
  const __bf16* arow = ao + (size_t)(b * S + m0 + ln) * AD;

  auto stage = [&](int buf, int k0) {
#pragma unroll
    for (int i = 0; i < 2; ++i) {
      int idx = tid + i * 128;
      int row = idx >> 2, c2 = idx & 3;
      async_load16(W + (size_t)(n0 + row) * AD + k0 + c2 * 8,
                   &wbuf[buf][row][c2 * 8]);
    }
  };

  v8f acc[4];
#pragma unroll
  for (int c = 0; c < 4; ++c) acc[c] = zero8();

  stage(0, 0);
  for (int k0 = 0; k0 < AD; k0 += 32) {
    const int cur = (k0 >> 5) & 1;
    if (k0 + 32 < AD) {
      stage(cur ^ 1, k0 + 32);
      asm volatile("s_wait_asynccnt 0x2" ::: "memory");
    } else {
      asm volatile("s_wait_asynccnt 0x0" ::: "memory");
    }
    __syncthreads();
    v16bf af = pack16(arow + k0 + g * 8, arow + k0 + 16 + g * 8);
#pragma unroll
    for (int c = 0; c < 4; ++c) {
      const __bf16* wr = &wbuf[cur][c * 16 + ln][8 * g];
      acc[c] = wmma_bf16(af, pack16(wr, wr + 16), acc[c]);
    }
    __syncthreads();
  }
#pragma unroll
  for (int r = 0; r < 8; ++r) {
    int s = m0 + r + 8 * g;
    float mv = (s < P) ? (nmask[b * MAXN + s] ? 1.f : 0.f)
                       : (tmask[b * T + (s - P)] ? 1.f : 0.f);
    size_t base = (s < P) ? (size_t)(b * P + s) * DIM
                          : (size_t)B * P * DIM + (size_t)(b * T + (s - P)) * DIM;
#pragma unroll
    for (int c = 0; c < 4; ++c) {
      int n = n0 + c * 16 + ln;
      out[base + n] = (acc[c][r] + bias[n]) * mv;
    }
  }
}

extern "C" void kernel_launch(void* const* d_in, const int* in_sizes, int n_in,
                              void* d_out, int out_size, void* d_ws, size_t ws_size,
                              hipStream_t stream) {
  using namespace cfg;
  (void)in_sizes; (void)n_in; (void)out_size; (void)ws_size;
  const float* pool        = (const float*)d_in[0];
  const float* x           = (const float*)d_in[1];
  const int* regions       = (const int*)d_in[2];
  const unsigned char* t_mask = (const unsigned char*)d_in[3];
  const unsigned char* n_mask = (const unsigned char*)d_in[4];
  const unsigned char* a_mask = (const unsigned char*)d_in[5];
  const float* pool_norm_w = (const float*)d_in[6];
  const float* x_norm_w    = (const float*)d_in[7];
  const float* pool_qkv_w  = (const float*)d_in[8];
  const float* pool_qkv_b  = (const float*)d_in[9];
  const float* x_qkv_w     = (const float*)d_in[10];
  const float* x_qkv_b     = (const float*)d_in[11];
  const float* pool_q_norm = (const float*)d_in[12];
  const float* pool_k_norm = (const float*)d_in[13];
  const float* x_q_norm    = (const float*)d_in[14];
  const float* x_k_norm    = (const float*)d_in[15];
  const float* pool_out_w  = (const float*)d_in[16];
  const float* pool_out_b  = (const float*)d_in[17];
  const float* x_out_w     = (const float*)d_in[18];
  const float* x_out_b     = (const float*)d_in[19];

  char* ws = (char*)d_ws;
  __bf16* wqp = (__bf16*)(ws + OFF_WQP);
  __bf16* wqx = (__bf16*)(ws + OFF_WQX);
  __bf16* wop = (__bf16*)(ws + OFF_WOP);
  __bf16* wox = (__bf16*)(ws + OFF_WOX);
  __bf16* an  = (__bf16*)(ws + OFF_AN);
  __bf16* qh  = (__bf16*)(ws + OFF_QB);
  __bf16* khb = (__bf16*)(ws + OFF_KB);
  __bf16* vtb = (__bf16*)(ws + OFF_VT);
  __bf16* ao  = (__bf16*)(ws + OFF_AO);
  int* lp     = (int*)(ws + OFF_LP);

  {
    size_t total = 2 * (size_t)3 * AD * DIM + 2 * (size_t)DIM * AD;
    int blocks = (int)((total + 255) / 256);
    convert_weights<<<blocks, 256, 0, stream>>>(pool_qkv_w, x_qkv_w, pool_out_w,
                                                x_out_w, wqp, wqx, wop, wox);
  }
  local_pos<<<B, 32, 0, stream>>>(regions, lp);
  rmsnorm_tokens<<<B * S, 256, 0, stream>>>(pool, x, pool_norm_w, x_norm_w, an);
  qkv_fused<<<dim3(3 * AD / 64, S / 16 / 4, B), 128, 0, stream>>>(
      an, wqp, wqx, pool_qkv_b, x_qkv_b, pool_q_norm, pool_k_norm, x_q_norm,
      x_k_norm, lp, qh, khb, vtb);
  flash_attn<<<dim3(S / 16, H, B), 32, 0, stream>>>(qh, khb, vtb, a_mask, ao);
  out_proj<<<dim3(DIM / 64, S / 16 / 4, B), 128, 0, stream>>>(
      ao, wop, wox, pool_out_b, x_out_b, n_mask, t_mask, (float*)d_out);
}
